// MultiHeadAttentionWithPosEncoding_25821343384321
// MI455X (gfx1250) — compile-verified, matrix operand order checked by reading, not run on hardware
//
#include <hip/hip_runtime.h>
#include <hip/hip_bf16.h>

// ---------------------------------------------------------------------------
// MultiHeadAttentionWithPosEncoding for gfx1250 (MI455X), bf16 WMMA pipeline.
//   Kernel 1: fused 1x1-conv (QKV GEMM) + bias + 2x2 maxpool + head-split
//             scatter; pos-encoding and 1/sqrt(d) folded into q; V stored
//             pre-transposed [b,h,dim,key] so kernel 2 can async-copy it.
//   Kernel 2: flash attention (online softmax); K/V staged with
//             global_load_async_to_lds_b128 (ASYNCcnt); WMMA bf16 for both
//             QK^T and P@V; transpose-store to [B,C,Hp,Wp] f32 output.
// ---------------------------------------------------------------------------

typedef __attribute__((ext_vector_type(16))) __bf16 v16bf;
typedef __attribute__((ext_vector_type(8)))  float  v8f;

union FragBF { unsigned u[8]; v16bf v; };

__device__ inline __bf16 f2bf(float f) {
  union { float f; unsigned u; } a; a.f = f;
  unsigned r = a.u + 0x7FFFu + ((a.u >> 16) & 1u);   // round-to-nearest-even
  unsigned short h = (unsigned short)(r >> 16);
  __bf16 o; __builtin_memcpy(&o, &h, 2); return o;
}

// Per-lane K-offset for 16-bit A/B fragments (ISA 7.12.2):
// lanes 0-15: VGPR r -> K = {2r,2r+1} (r<4) / {16+2(r-4),...} (r>=4)
// lanes 16-31: same +8.
__device__ inline int frag_koff(int r, int half) {
  return ((r >> 2) << 4) + (half << 3) + ((r & 3) << 1);
}

// Async cache->LDS copy, 16 bytes per enabled lane (gfx1250 ASYNCcnt path).
// VDST VGPR carries the 32-bit LDS byte address; VADDR the 64-bit global addr.
__device__ inline void async_copy_b128(const void* gsrc, void* ldst) {
  unsigned lds_addr = (unsigned)(unsigned long long)ldst;  // low 32 = LDS offset
  asm volatile("global_load_async_to_lds_b128 %0, %1, off"
               :: "v"(lds_addr), "v"(gsrc) : "memory");
}
__device__ inline void wait_asynccnt0() {
  asm volatile("s_wait_asynccnt 0x0" ::: "memory");
}

// ---------------------------------------------------------------------------
// Kernel 1: QKV projection + pool.
// grid = (12 channel-blocks of 64, Hp=32, B=16), block = 256 (8 waves).
// Each workgroup: Y[64 x 128] = W[64 x 256] * X[256 x 128] (two image rows),
// then pools to 64 x 32 and scatters.
// ---------------------------------------------------------------------------
__global__ __launch_bounds__(256)
void qkv_pool_kernel(const float* __restrict__ x, const float* __restrict__ w,
                     const float* __restrict__ bias, const float* __restrict__ pos,
                     __bf16* __restrict__ qb, __bf16* __restrict__ kb,
                     __bf16* __restrict__ vtb)
{
  __shared__ __bf16 lds_w[64 * 32];    // [m][k]
  __shared__ __bf16 lds_x[128 * 32];   // [s][k]  (K-contiguous for B frags)
  __shared__ float  lds_y[64 * 128];   // conv result before pooling

  const int tid  = threadIdx.x;
  const int ob   = blockIdx.x;   // 64-channel block: 0..11
  const int hp   = blockIdx.y;   // pooled row: 0..31
  const int b    = blockIdx.z;   // batch
  const int o0   = ob * 64;
  const int s0   = hp * 128;     // rows 2hp,2hp+1 are contiguous 128 floats

  const int wv   = tid >> 5;
  const int lane = tid & 31;
  const int ml   = lane & 15;
  const int half = (lane >> 4) & 1;
  const int mt   = wv & 3;       // M tile (16 rows) of this wave
  const int sg   = wv >> 2;      // S-tile group (4 tiles each)

  v8f acc[4];
  for (int i = 0; i < 4; i++)
    for (int r = 0; r < 8; r++) acc[i][r] = 0.f;

  int koff[8];
  for (int r = 0; r < 8; r++) koff[r] = frag_koff(r, half);

  for (int k0 = 0; k0 < 256; k0 += 32) {
    // prefetch next X tile into GL2 while we convert/compute this one
    if (k0 < 224)
      __builtin_prefetch(&x[((size_t)(b * 256 + k0 + 32 + (tid >> 3))) * 4096 + s0], 0, 1);

    // stage W tile 64x32 (8 elems/thread, coalesced along K)
    for (int i = 0; i < 8; i++) {
      int e = tid + i * 256;
      int m = e >> 5, kk = e & 31;
      lds_w[e] = f2bf(w[(o0 + m) * 256 + k0 + kk]);
    }
    // stage X tile [s][k] (16 elems/thread, coalesced along s)
    for (int i = 0; i < 16; i++) {
      int e = tid + i * 256;
      int s = e & 127, kk = e >> 7;
      lds_x[s * 32 + kk] = f2bf(x[((size_t)(b * 256 + k0 + kk)) * 4096 + s0 + s]);
    }
    __syncthreads();

    FragBF a;
    for (int r = 0; r < 8; r++)
      a.u[r] = *(const unsigned*)(&lds_w[(mt * 16 + ml) * 32 + koff[r]]);
    for (int i = 0; i < 4; i++) {
      int st = sg * 4 + i;
      FragBF bf;
      for (int r = 0; r < 8; r++)
        bf.u[r] = *(const unsigned*)(&lds_x[(st * 16 + ml) * 32 + koff[r]]);
      acc[i] = __builtin_amdgcn_wmma_f32_16x16x32_bf16(
          false, a.v, false, bf.v, (short)0, acc[i], false, false);
    }
    __syncthreads();
  }

  // park conv result in LDS (C layout: lanes0-15 -> M=r, lanes16-31 -> M=8+r)
  for (int i = 0; i < 4; i++) {
    int st = sg * 4 + i;
    for (int r = 0; r < 8; r++)
      lds_y[(mt * 16 + half * 8 + r) * 128 + st * 16 + ml] = acc[i][r];
  }
  __syncthreads();

  // 2x2 maxpool + bias + head-split scatter (bias commutes with max)
  const int t = o0 >> 8;                       // 0=q, 1=k, 2=v
  __bf16* dst = (t == 0) ? qb : (t == 1) ? kb : vtb;
  const float scale = 0.17677669529663687f;    // 1/sqrt(32)
  for (int i = 0; i < 8; i++) {
    int e = tid + i * 256;
    int m = e >> 5, wp = e & 31;
    const float* row = &lds_y[m * 128];
    float v0 = fmaxf(row[2 * wp],      row[2 * wp + 1]);
    float v1 = fmaxf(row[64 + 2 * wp], row[64 + 2 * wp + 1]);
    float val = fmaxf(v0, v1) + bias[o0 + m];
    int o    = o0 + m;
    int rem  = o & 255;
    int head = rem >> 5, dim = rem & 31;
    int n    = hp * 32 + wp;
    size_t idx;
    if (t == 0) {
      val = (val + pos[n * 32 + dim]) * scale;               // fold pos + scale
      idx = ((size_t)(b * 8 + head) * 1024 + n) * 32 + dim;  // [b,h,n,d]
    } else if (t == 1) {
      idx = ((size_t)(b * 8 + head) * 1024 + n) * 32 + dim;  // [b,h,n,d]
    } else {
      idx = ((size_t)(b * 8 + head) * 32 + dim) * 1024 + n;  // V^T: [b,h,d,n]
    }
    dst[idx] = f2bf(val);
  }
}

// ---------------------------------------------------------------------------
// Kernel 2: flash attention.
// grid = (8 row-blocks of 128, heads=8, B=16), block = 256 (8 waves x 16 rows).
// d=32 == one WMMA K-step for QK^T; P@V splits K=64 into two K=32 steps.
// K and V^T blocks staged with async cache->LDS b128 copies.
// ---------------------------------------------------------------------------
__global__ __launch_bounds__(256)
void attn_kernel(const __bf16* __restrict__ qb, const __bf16* __restrict__ kb,
                 const __bf16* __restrict__ vtb, float* __restrict__ out)
{
  __shared__ __bf16 lds_k[64 * 32];       // [key][dim]
  __shared__ __bf16 lds_vt[32 * 64];      // [dim][key] (K-pairs contiguous)
  __shared__ __bf16 lds_p[8][16 * 64];    // per-wave P relayout slab

  const int tid  = threadIdx.x;
  const int rb   = blockIdx.x;
  const int h    = blockIdx.y;
  const int b    = blockIdx.z;

  const int wv   = tid >> 5;
  const int lane = tid & 31;
  const int ml   = lane & 15;
  const int half = (lane >> 4) & 1;

  const size_t bh = (size_t)(b * 8 + h) * 1024 * 32;
  const __bf16* q  = qb  + bh;
  const __bf16* k  = kb  + bh;
  const __bf16* vt = vtb + bh;   // [dim][key] layout, 1024 keys per dim row

  int koff[8];
  for (int r = 0; r < 8; r++) koff[r] = frag_koff(r, half);

  const int qrow0 = rb * 128 + wv * 16;

  // Q fragment straight from global (q already has pos + 1/sqrt(d) folded)
  FragBF qf;
  for (int r = 0; r < 8; r++)
    qf.u[r] = *(const unsigned*)(q + (qrow0 + ml) * 32 + koff[r]);

  v8f acc0, acc1;
  float mrun[8], lrun[8];
  for (int r = 0; r < 8; r++) { acc0[r] = 0.f; acc1[r] = 0.f; mrun[r] = -1e30f; lrun[r] = 0.f; }

  const int vdim = tid >> 3, vchunk = tid & 7;   // V^T staging coords

  for (int j = 0; j < 16; j++) {
    const int key0 = j * 64;

    // async cache->LDS staging: K block 4KB + V^T block 4KB, b128/lane
    async_copy_b128(k + key0 * 32 + tid * 8, &lds_k[tid * 8]);
    async_copy_b128(vt + vdim * 1024 + key0 + vchunk * 8,
                    &lds_vt[vdim * 64 + vchunk * 8]);
    if (j < 15) {   // prefetch next K block into GL2
      __builtin_prefetch(k + (key0 + 64) * 32 + tid * 8, 0, 1);
      __builtin_prefetch(vt + vdim * 1024 + key0 + 64 + vchunk * 8, 0, 1);
    }
    wait_asynccnt0();
    __syncthreads();

    // S = Q K^T : 4 x 16-col subtiles
    v8f s[4];
    for (int sub = 0; sub < 4; sub++) {
      FragBF bf;
      for (int r = 0; r < 8; r++)
        bf.u[r] = *(const unsigned*)(&lds_k[(sub * 16 + ml) * 32 + koff[r]]);
      v8f z; for (int r = 0; r < 8; r++) z[r] = 0.f;
      s[sub] = __builtin_amdgcn_wmma_f32_16x16x32_bf16(
          false, qf.v, false, bf.v, (short)0, z, false, false);
    }

    // online softmax; row (r,half) reduces across its 16-lane group
    for (int r = 0; r < 8; r++) {
      float v0 = s[0][r], v1 = s[1][r], v2 = s[2][r], v3 = s[3][r];
      float bm = fmaxf(fmaxf(v0, v1), fmaxf(v2, v3));
      bm = fmaxf(bm, __shfl_xor(bm, 1, 32));
      bm = fmaxf(bm, __shfl_xor(bm, 2, 32));
      bm = fmaxf(bm, __shfl_xor(bm, 4, 32));
      bm = fmaxf(bm, __shfl_xor(bm, 8, 32));
      float mnew = fmaxf(mrun[r], bm);
      float corr = __expf(mrun[r] - mnew);
      float p0 = __expf(v0 - mnew), p1 = __expf(v1 - mnew);
      float p2 = __expf(v2 - mnew), p3 = __expf(v3 - mnew);
      float rs = p0 + p1 + p2 + p3;
      rs += __shfl_xor(rs, 1, 32);
      rs += __shfl_xor(rs, 2, 32);
      rs += __shfl_xor(rs, 4, 32);
      rs += __shfl_xor(rs, 8, 32);
      lrun[r] = lrun[r] * corr + rs;
      mrun[r] = mnew;
      acc0[r] *= corr; acc1[r] *= corr;
      // C-frag -> A-frag relayout via per-wave LDS slab
      int prow = (r + half * 8) * 64;
      lds_p[wv][prow +  0 + ml] = f2bf(p0);
      lds_p[wv][prow + 16 + ml] = f2bf(p1);
      lds_p[wv][prow + 32 + ml] = f2bf(p2);
      lds_p[wv][prow + 48 + ml] = f2bf(p3);
    }

    // O += P @ V : two K=32 chunks x two 16-dim tiles
    for (int c = 0; c < 2; c++) {
      FragBF pf;
      for (int r = 0; r < 8; r++)
        pf.u[r] = *(const unsigned*)(&lds_p[wv][ml * 64 + c * 32 + koff[r]]);
      FragBF vf0, vf1;
      for (int r = 0; r < 8; r++) {
        vf0.u[r] = *(const unsigned*)(&lds_vt[ml * 64        + c * 32 + koff[r]]);
        vf1.u[r] = *(const unsigned*)(&lds_vt[(16 + ml) * 64 + c * 32 + koff[r]]);
      }
      acc0 = __builtin_amdgcn_wmma_f32_16x16x32_bf16(
          false, pf.v, false, vf0.v, (short)0, acc0, false, false);
      acc1 = __builtin_amdgcn_wmma_f32_16x16x32_bf16(
          false, pf.v, false, vf1.v, (short)0, acc1, false, false);
    }
    __syncthreads();
  }

  // normalize + transpose-store to [B, C=head*32+dim, Hp, Wp]
  for (int r = 0; r < 8; r++) {
    int n  = qrow0 + r + half * 8;
    int hp = n >> 5, wp = n & 31;
    float inv = 1.0f / lrun[r];
    int c0 = h * 32 + ml;
    int c1 = h * 32 + 16 + ml;
    out[(((size_t)b * 256 + c0) * 32 + hp) * 32 + wp] = acc0[r] * inv;
    out[(((size_t)b * 256 + c1) * 32 + hp) * 32 + wp] = acc1[r] * inv;
  }
}

// ---------------------------------------------------------------------------
extern "C" void kernel_launch(void* const* d_in, const int* in_sizes, int n_in,
                              void* d_out, int out_size, void* d_ws, size_t ws_size,
                              hipStream_t stream) {
  const float* x    = (const float*)d_in[0];
  const float* w    = (const float*)d_in[1];
  const float* bias = (const float*)d_in[2];
  const float* pos  = (const float*)d_in[3];

  const size_t per = (size_t)16 * 8 * 1024 * 32;   // elements per q/k/v buffer
  __bf16* qb  = (__bf16*)d_ws;
  __bf16* kb  = qb + per;
  __bf16* vtb = kb + per;

  qkv_pool_kernel<<<dim3(12, 32, 16), 256, 0, stream>>>(x, w, bias, pos, qb, kb, vtb);
  attn_kernel<<<dim3(8, 8, 16), 256, 0, stream>>>(qb, kb, vtb, (float*)d_out);
}
